// MoE_41970420418120
// MI455X (gfx1250) — matrix operand order repaired and checked by reading, not compiled, not run on hardware
//
#include <hip/hip_runtime.h>
#include <hip/hip_bf16.h>

// ---------------- problem constants (match reference) ----------------
constexpr int NT  = 8192;          // tokens = B*S
constexpr int NH  = 2048;          // hidden
constexpr int NE  = 8;             // experts
constexpr int NI  = 1408;          // routed intermediate (shared expert = 2 halves of NI)
constexpr int NSI = 2816;          // shared intermediate
constexpr int MT  = 32;            // token tile (2 WMMA M-fragments)
constexpr int SA_STRIDE = NH + 8;  // padded LDS strides
constexpr int H1_STRIDE = NI + 8;

typedef __attribute__((ext_vector_type(16))) __bf16 v16bf;
typedef __attribute__((ext_vector_type(8)))  float  v8f;

union BFPack { v16bf v; unsigned u[8]; };

// two f32 -> packed bf16 dword: round-to-nearest via +0x8000, pack via v_perm_b32
__device__ __forceinline__ unsigned cvt2bf(float lo, float hi) {
    unsigned ul = __builtin_bit_cast(unsigned, lo) + 0x8000u;
    unsigned uh = __builtin_bit_cast(unsigned, hi) + 0x8000u;
    return __builtin_amdgcn_perm(uh, ul, 0x07060302u);   // {uh[31:16], ul[31:16]}
}

__device__ __forceinline__ unsigned short cvt1bf(float f) {
    return (unsigned short)((__builtin_bit_cast(unsigned, f) + 0x8000u) >> 16);
}

__device__ __forceinline__ float fast_rcp(float v) {
#if __has_builtin(__builtin_amdgcn_rcpf)
    return __builtin_amdgcn_rcpf(v);
#else
    return 1.f / v;
#endif
}

__device__ __forceinline__ v8f wmma_bf16(v16bf a, v16bf b, v8f c) {
    return __builtin_amdgcn_wmma_f32_16x16x32_bf16(
        false, a, false, b, (short)0, c, false, false);
}

// A fragment (16x32 bf16) from padded row-major LDS.
// Layout used here (contiguous k, same as load_Bfrag): lane M = lane&15, half = lane>>4; element p -> K =
// 16*half + p.  K-consecutive pairs -> 8 dword LDS loads.
__device__ __forceinline__ v16bf lds_A_frag(const unsigned short* base, int stride,
                                            int k0, int lane) {
    const int M = lane & 15, half = lane >> 4;
    const int be = M * stride + k0 + 16 * half;
    BFPack A;
#pragma unroll
    for (int q = 0; q < 8; ++q) {
        int kb = 2 * q;
        A.u[q] = *(const unsigned*)(base + be + kb);
    }
    return A.v;
}

// B fragment: B[k][n] = W[n0+n][k0+k], row-major W[.][ld].
// bf16 weights: 16 K-consecutive bf16 per lane == contiguous k order of lds_A_frag, 2x b128, no VALU.
__device__ __forceinline__ v16bf load_Bfrag(const unsigned short* __restrict__ base,
                                            int ld, int n0, int k0, int lane) {
    const int n  = n0 + (lane & 15);
    const int kb = k0 + ((lane >> 4) << 4);
    const uint4* p = (const uint4*)(base + (size_t)n * ld + kb);
    uint4 q0 = p[0], q1 = p[1];
    BFPack B;
    B.u[0] = q0.x; B.u[1] = q0.y; B.u[2] = q0.z; B.u[3] = q0.w;
    B.u[4] = q1.x; B.u[5] = q1.y; B.u[6] = q1.z; B.u[7] = q1.w;
    return B.v;
}

// f32-weight fallback: 4x b128 + add/perm conversion
__device__ __forceinline__ v16bf load_Bfrag(const float* __restrict__ base,
                                            int ld, int n0, int k0, int lane) {
    const int n  = n0 + (lane & 15);
    const int kb = k0 + ((lane >> 4) << 4);
    const float4* p = (const float4*)(base + (size_t)n * ld + kb);
    BFPack B;
#pragma unroll
    for (int j = 0; j < 4; ++j) {
        float4 f = p[j];
        B.u[2 * j]     = cvt2bf(f.x, f.y);
        B.u[2 * j + 1] = cvt2bf(f.z, f.w);
    }
    return B.v;
}

// ---------------- kernel: f32 -> bf16 weight conversion (grid-stride) -------
__global__ __launch_bounds__(256) void cvt_bf16_kernel(
    const float* __restrict__ src, unsigned short* __restrict__ dst, long n) {
    long i = ((long)blockIdx.x * 256 + threadIdx.x) * 8;
    const long stride = (long)gridDim.x * 256 * 8;
    for (; i < n; i += stride) {
        const float4* s = (const float4*)(src + i);
        float4 f0 = s[0], f1 = s[1];
        uint4 w4;
        w4.x = cvt2bf(f0.x, f0.y); w4.y = cvt2bf(f0.z, f0.w);
        w4.z = cvt2bf(f1.x, f1.y); w4.w = cvt2bf(f1.z, f1.w);
        *(uint4*)(dst + i) = w4;
    }
}

// ---------------- kernel: zero expert counters ----------------
__global__ void moe_init_kernel(int* __restrict__ cnt) {
    if (threadIdx.x < NE) cnt[threadIdx.x] = 0;
}

// ---------------- kernel: gate + top-2 + compaction ----------------
__global__ __launch_bounds__(256) void moe_gate_kernel(
    const float* __restrict__ x, const float* __restrict__ gw,
    int* __restrict__ cnt, int* __restrict__ listIdx, float* __restrict__ listW) {
    const int wave = threadIdx.x >> 5, lane = threadIdx.x & 31;
    const int t = blockIdx.x * 8 + wave;
    float acc[NE];
#pragma unroll
    for (int e = 0; e < NE; ++e) acc[e] = 0.f;
    for (int c = lane; c < NH; c += 32) {
        float xv = x[(size_t)t * NH + c];
#pragma unroll
        for (int e = 0; e < NE; ++e) acc[e] += xv * gw[e * NH + c];
    }
#pragma unroll
    for (int e = 0; e < NE; ++e)
        for (int off = 16; off > 0; off >>= 1)
            acc[e] += __shfl_xor(acc[e], off, 32);
    if (lane == 0) {
        float m = acc[0];
#pragma unroll
        for (int e = 1; e < NE; ++e) m = fmaxf(m, acc[e]);
        float p[NE], s = 0.f;
#pragma unroll
        for (int e = 0; e < NE; ++e) { p[e] = __expf(acc[e] - m); s += p[e]; }
        float inv = fast_rcp(s);
#pragma unroll
        for (int e = 0; e < NE; ++e) p[e] *= inv;
        int e1 = 0;
#pragma unroll
        for (int e = 1; e < NE; ++e) if (p[e] > p[e1]) e1 = e;
        int e2 = (e1 == 0) ? 1 : 0;
#pragma unroll
        for (int e = 0; e < NE; ++e) if (e != e1 && p[e] > p[e2]) e2 = e;
        float tot = fast_rcp(p[e1] + p[e2] + 1e-20f);
        int s1 = atomicAdd(&cnt[e1], 1);
        listIdx[(size_t)e1 * NT + s1] = t; listW[(size_t)e1 * NT + s1] = p[e1] * tot;
        int s2 = atomicAdd(&cnt[e2], 1);
        listIdx[(size_t)e2 * NT + s2] = t; listW[(size_t)e2 * NT + s2] = p[e2] * tot;
    }
}

// ---------------- kernel: fused swiglu MLP, 32-token tile ----------------
// WT = unsigned short (pre-converted bf16 weights) or float (fallback).
// routed==1: job = routed expert (gathered tokens, weighted atomic scatter).
// routed==0: job = shared-expert half v (all tokens, weight 1, atomic scatter).
template <typename WT>
__global__ __launch_bounds__(512) void moe_mlp_kernel(
    const float* __restrict__ x, const WT* __restrict__ wgBase,
    const WT* __restrict__ wdBase, float* __restrict__ out,
    const int* __restrict__ cnt, const int* __restrict__ listIdx,
    const float* __restrict__ listW, int tilesPerJob, int routed) {
    const int tid = threadIdx.x;
    const int wave = tid >> 5, lane = tid & 31;
    const int job = blockIdx.x / tilesPerJob;
    const int tile = blockIdx.x - job * tilesPerJob;

    const WT *wgGate, *wgProj, *wd;
    int ldwd, n_e;
    if (routed) {
        n_e = cnt[job];
        if (tile * MT >= n_e) return;                       // uniform early exit
        wgGate = wgBase + (size_t)job * (2 * NI) * NH;
        wgProj = wgGate + (size_t)NI * NH;
        wd     = wdBase + (size_t)job * NH * NI;
        ldwd   = NI;
    } else {
        n_e = NT;
        wgGate = wgBase + (size_t)job * NI * NH;
        wgProj = wgBase + ((size_t)NSI + (size_t)job * NI) * NH;
        wd     = wdBase + (size_t)job * NI;
        ldwd   = NSI;
    }

    extern __shared__ char smem[];
    unsigned short* sA  = (unsigned short*)smem;                         // MT x SA_STRIDE
    unsigned short* sH1 = (unsigned short*)(smem + (size_t)MT * SA_STRIDE * 2);
    int*   sIdx = (int*)(smem + (size_t)MT * SA_STRIDE * 2 + (size_t)MT * H1_STRIDE * 2);
    float* sW   = (float*)(sIdx + MT);

    if (tid < MT) {
        int gi = tile * MT + tid, idx; float w;
        if (routed) {
            if (gi < n_e) { idx = listIdx[(size_t)job * NT + gi];
                            w  = listW[(size_t)job * NT + gi]; }
            else          { idx = 0; w = 0.f; }             // padded row
        } else            { idx = gi; w = 1.f; }
        sIdx[tid] = idx; sW[tid] = w;
    }
    __syncthreads();

    // stage gathered A (MT x NH) as bf16
    for (int i = tid * 8; i < MT * NH; i += 512 * 8) {
        int r = i >> 11, c = i & (NH - 1);
        const float4* src = (const float4*)(x + (size_t)sIdx[r] * NH + c);
        float4 f0 = src[0], f1 = src[1];
        uint4 w4;
        w4.x = cvt2bf(f0.x, f0.y); w4.y = cvt2bf(f0.z, f0.w);
        w4.z = cvt2bf(f1.x, f1.y); w4.w = cvt2bf(f1.z, f1.w);
        *(uint4*)(sA + r * SA_STRIDE + c) = w4;
    }
    __syncthreads();

    // ---- GEMM1 (+swiglu): each wave: 32x16 gate tile + 32x16 proj tile ----
    for (int pt = wave; pt < (NI >> 4); pt += 16) {
        const int n0 = pt << 4;
        v8f aG0 = {}, aG1 = {}, aP0 = {}, aP1 = {};
#pragma unroll 2
        for (int k0 = 0; k0 < NH; k0 += 32) {
            v16bf bG = load_Bfrag(wgGate, NH, n0, k0, lane);
            v16bf bP = load_Bfrag(wgProj, NH, n0, k0, lane);
            v16bf a0 = lds_A_frag(sA, SA_STRIDE, k0, lane);
            v16bf a1 = lds_A_frag(sA + 16 * SA_STRIDE, SA_STRIDE, k0, lane);
            aG0 = wmma_bf16(a0, bG, aG0);
            aG1 = wmma_bf16(a1, bG, aG1);
            aP0 = wmma_bf16(a0, bP, aP0);
            aP1 = wmma_bf16(a1, bP, aP1);
        }
        const int col = n0 + (lane & 15);
        const int M0 = (lane >> 4) * 8;
#pragma unroll
        for (int v = 0; v < 8; ++v) {
            float g0 = aG0[v];
            float h0 = g0 * fast_rcp(1.f + __expf(-g0)) * aP0[v];   // silu(g)*p
            sH1[(M0 + v) * H1_STRIDE + col] = cvt1bf(h0);
            float g1 = aG1[v];
            float h1 = g1 * fast_rcp(1.f + __expf(-g1)) * aP1[v];
            sH1[(16 + M0 + v) * H1_STRIDE + col] = cvt1bf(h1);
        }
    }
    __syncthreads();

    // ---- GEMM2: out_tile += w .* (h1 @ wd.T) via f32 atomics ----
    for (int nt = wave; nt < (NH >> 4); nt += 16) {
        const int n0 = nt << 4;
        v8f c0 = {}, c1 = {};
#pragma unroll 2
        for (int k0 = 0; k0 < NI; k0 += 32) {
            v16bf b  = load_Bfrag(wd, ldwd, n0, k0, lane);
            v16bf a0 = lds_A_frag(sH1, H1_STRIDE, k0, lane);
            v16bf a1 = lds_A_frag(sH1 + 16 * H1_STRIDE, H1_STRIDE, k0, lane);
            c0 = wmma_bf16(a0, b, c0);
            c1 = wmma_bf16(a1, b, c1);
        }
        const int col = n0 + (lane & 15);
        const int M0 = (lane >> 4) * 8;
#pragma unroll
        for (int v = 0; v < 8; ++v) {
            float w0 = sW[M0 + v];
            if (w0 != 0.f)
                atomicAdd(out + (size_t)sIdx[M0 + v] * NH + col, w0 * c0[v]);
            float w1 = sW[16 + M0 + v];
            if (w1 != 0.f)
                atomicAdd(out + (size_t)sIdx[16 + M0 + v] * NH + col, w1 * c1[v]);
        }
    }
}

// ---------------- host launch ----------------
extern "C" void kernel_launch(void* const* d_in, const int* in_sizes, int n_in,
                              void* d_out, int out_size, void* d_ws, size_t ws_size,
                              hipStream_t stream) {
    const float* x         = (const float*)d_in[0];
    const float* gate_w    = (const float*)d_in[1];
    const float* expert_wg = (const float*)d_in[2];
    const float* expert_wd = (const float*)d_in[3];
    const float* shared_wg = (const float*)d_in[4];
    const float* shared_wd = (const float*)d_in[5];
    float* out = (float*)d_out;

    char* ws = (char*)d_ws;
    int*   cnt     = (int*)ws;                               // 64 B
    int*   listIdx = (int*)(ws + 64);                        // NE*NT ints
    float* listW   = (float*)(ws + 64 + (size_t)NE * NT * 4);
    const size_t wOff = ((64 + (size_t)NE * NT * 8) + 255) & ~(size_t)255;

    const long nEg = (long)NE * 2 * NI * NH;    // expert gate+proj
    const long nEd = (long)NE * NH * NI;        // expert down
    const long nSg = (long)2 * NSI * NH;        // shared gate+proj
    const long nSd = (long)NH * NSI;            // shared down
    unsigned short* bEg = (unsigned short*)(ws + wOff);
    unsigned short* bEd = bEg + nEg;
    unsigned short* bSg = bEd + nEd;
    unsigned short* bSd = bSg + nSg;
    const size_t need = wOff + (size_t)(nEg + nEd + nSg + nSd) * 2;
    const bool pre = ws_size >= need;           // deterministic per-harness choice

    const size_t shmem = (size_t)MT * SA_STRIDE * 2 + (size_t)MT * H1_STRIDE * 2
                       + MT * (sizeof(int) + sizeof(float));   // ~217 KB

    (void)hipFuncSetAttribute((const void*)moe_mlp_kernel<unsigned short>,
                              hipFuncAttributeMaxDynamicSharedMemorySize, 224 * 1024);
    (void)hipFuncSetAttribute((const void*)moe_mlp_kernel<float>,
                              hipFuncAttributeMaxDynamicSharedMemorySize, 224 * 1024);

    (void)hipMemsetAsync(d_out, 0, (size_t)out_size * sizeof(float), stream);
    moe_init_kernel<<<1, 32, 0, stream>>>(cnt);
    moe_gate_kernel<<<NT / 8, 256, 0, stream>>>(x, gate_w, cnt, listIdx, listW);

    if (pre) {
        cvt_bf16_kernel<<<4096, 256, 0, stream>>>(expert_wg, bEg, nEg);
        cvt_bf16_kernel<<<4096, 256, 0, stream>>>(expert_wd, bEd, nEd);
        cvt_bf16_kernel<<<2048, 256, 0, stream>>>(shared_wg, bSg, nSg);
        cvt_bf16_kernel<<<1024, 256, 0, stream>>>(shared_wd, bSd, nSd);
        moe_mlp_kernel<unsigned short><<<2 * (NT / MT), 512, shmem, stream>>>(
            x, bSg, bSd, out, nullptr, nullptr, nullptr, NT / MT, 0);
        moe_mlp_kernel<unsigned short><<<NE * (NT / MT), 512, shmem, stream>>>(
            x, bEg, bEd, out, cnt, listIdx, listW, NT / MT, 1);
    } else {
        moe_mlp_kernel<float><<<2 * (NT / MT), 512, shmem, stream>>>(
            x, shared_wg, shared_wd, out, nullptr, nullptr, nullptr, NT / MT, 0);
        moe_mlp_kernel<float><<<NE * (NT / MT), 512, shmem, stream>>>(
            x, expert_wg, expert_wd, out, cnt, listIdx, listW, NT / MT, 1);
    }
}
